// PrimateReachSNN_15831249453621
// MI455X (gfx1250) — 8-bit matrix operand order repaired and checked by reading, compiled, not run on hardware
//
#include <hip/hip_runtime.h>


typedef int   v16i __attribute__((ext_vector_type(16)));
typedef float v8f  __attribute__((ext_vector_type(8)));

#define B_TOT  512
#define T_TOT  500
#define NIN    96
#define HID    512
#define NOUT   2

#define BT     32
#define NWAVE  8
#define NTHR   (NWAVE * 32)

#define WP     528
#define SP     528
#define XP     144
#define VPF    516

#define WLDS_BYTES (HID * WP)
#define SLDS_BYTES (BT * SP)
#define XLDS_BYTES (BT * XP)
#define LDS_TOTAL  (WLDS_BYTES + SLDS_BYTES + XLDS_BYTES)

#define WSCALE 64.0f
#define INV_WS (1.0f / 64.0f)

__device__ __forceinline__ unsigned char f32_to_e4m3(float xf) {
    unsigned u = __float_as_uint(xf);
    unsigned s = (u >> 24) & 0x80u;
    float a = __uint_as_float(u & 0x7fffffffu);
    if (!(a < 448.0f)) return (unsigned char)(s | 0x7e);
    int e = (int)((u >> 23) & 0xff) - 127;
    if (e < -6) e = -6;
    float q = a * exp2f((float)(3 - e));
    int qi = (int)(q + 0.5f);
    if (qi >= 16) { qi = 8; ++e; }
    if (e > 8) return (unsigned char)(s | 0x7e);
    if (qi < 8) return (unsigned char)(s | (unsigned)qi);
    return (unsigned char)(s | (unsigned)((e + 7) << 3) | (unsigned)(qi - 8));
}

__global__ void __launch_bounds__(NTHR, 1)
adlif_scan_fp8_wmma(const float* __restrict__ x,
                    const float* __restrict__ W1,
                    const float* __restrict__ Wrec,
                    const float* __restrict__ W2,
                    const float* __restrict__ alpha,
                    const float* __restrict__ rho,
                    const float* __restrict__ beta_a,
                    float* __restrict__ out)
{
    extern __shared__ unsigned char smem[];
    unsigned char* __restrict__ wl = smem;
    unsigned char* __restrict__ sl = smem + WLDS_BYTES;
    unsigned char* __restrict__ xl = smem + WLDS_BYTES + SLDS_BYTES;

    const int tid  = threadIdx.x;
    const int wv   = tid >> 5;
    const int lane = tid & 31;
    const int half = lane >> 4;
    const int lm   = lane & 15;
    const int b0   = blockIdx.x * BT;

    for (int idx = tid; idx < HID * HID; idx += NTHR) {
        int h = idx >> 9, k = idx & (HID - 1);
        wl[h * WP + k] = f32_to_e4m3(Wrec[idx] * WSCALE);
    }
    for (int idx = tid; idx < SLDS_BYTES; idx += NTHR) sl[idx] = 0;
    for (int idx = tid; idx < XLDS_BYTES; idx += NTHR) xl[idx] = 0;

    v16i w1f[4];
#pragma unroll
    for (int nt = 0; nt < 4; ++nt) {
        const int h = (wv << 6) + (nt << 4) + lm;
#pragma unroll
        for (int vv = 0; vv < 16; ++vv) {
            const int kb = ((vv >> 2) << 5) + (half << 4) + ((vv & 3) << 2);
            unsigned word = 0;
#pragma unroll
            for (int b = 0; b < 4; ++b) {
                const int kk = kb + b;
                const float w = (kk < NIN) ? W1[h * NIN + kk] * WSCALE : 0.0f;
                word |= ((unsigned)f32_to_e4m3(w)) << (8 * b);
            }
            w1f[nt][vv] = (int)word;
        }
    }

    float al_[4], oa_[4], oas_[4], rh_[4], bt_[4];
#pragma unroll
    for (int nt = 0; nt < 4; ++nt) {
        const int hc = (wv << 6) + (nt << 4) + lm;
        const float a1 = alpha[hc];
        al_[nt] = a1; oa_[nt] = 1.0f - a1; oas_[nt] = (1.0f - a1) * INV_WS;
        rh_[nt] = rho[hc]; bt_[nt] = beta_a[hc];
    }

    float vst[2][32], ast[2][32];
    unsigned spkm[2] = {0u, 0u};
#pragma unroll
    for (int m = 0; m < 2; ++m)
#pragma unroll
        for (int i = 0; i < 32; ++i) { vst[m][i] = 0.0f; ast[m][i] = 0.0f; }

    __syncthreads();

#pragma unroll 1
    for (int t = 0; t < T_TOT; ++t) {
        {
            const float* xb = x + (size_t)b0 * T_TOT * NIN + (size_t)t * NIN;
            for (int idx = tid; idx < BT * NIN; idx += NTHR) {
                const int bb = idx / NIN, i = idx - bb * NIN;
                const float* p = xb + (size_t)bb * T_TOT * NIN + i;
                xl[bb * XP + i] = f32_to_e4m3(*p);
                if (t + 1 < T_TOT) __builtin_prefetch(p + NIN, 0, 0);
            }
        }
        __syncthreads();

        unsigned nmask[2] = {0u, 0u};

#pragma unroll
        for (int m = 0; m < 2; ++m) {
            v16i ax;
            {
                const unsigned char* base = xl + ((m << 4) + lm) * XP + (half << 4);
#pragma unroll
                for (int j = 0; j < 8; ++j) {
                    const unsigned long long q =
                        *(const unsigned long long*)(base + ((j >> 1) << 5) + ((j & 1) << 3));
                    ax[2 * j]     = (int)(unsigned)q;
                    ax[2 * j + 1] = (int)(unsigned)(q >> 32);
                }
            }
            v16i asf[4];
#pragma unroll
            for (int ks = 0; ks < 4; ++ks) {
                const unsigned char* base =
                    sl + ((m << 4) + lm) * SP + (ks << 7) + (half << 4);
#pragma unroll
                for (int j = 0; j < 8; ++j) {
                    const unsigned long long q =
                        *(const unsigned long long*)(base + ((j >> 1) << 5) + ((j & 1) << 3));
                    asf[ks][2 * j]     = (int)(unsigned)q;
                    asf[ks][2 * j + 1] = (int)(unsigned)(q >> 32);
                }
            }

#pragma unroll
            for (int nt = 0; nt < 4; ++nt) {
                const int hrow = (wv << 6) + (nt << 4) + lm;
                v8f c = {};
                c = __builtin_amdgcn_wmma_f32_16x16x128_fp8_fp8(
                        ax, w1f[nt], (short)0, c, false, false);
#pragma unroll
                for (int ks = 0; ks < 4; ++ks) {
                    v16i bw;
                    const unsigned char* base =
                        wl + hrow * WP + (ks << 7) + (half << 4);
#pragma unroll
                    for (int cc = 0; cc < 4; ++cc) {
                        const int4 d = *(const int4*)(base + (cc << 5));
                        bw[4 * cc + 0] = d.x; bw[4 * cc + 1] = d.y;
                        bw[4 * cc + 2] = d.z; bw[4 * cc + 3] = d.w;
                    }
                    c = __builtin_amdgcn_wmma_f32_16x16x128_fp8_fp8(
                            asf[ks], bw, (short)0, c, false, false);
                }
#pragma unroll
                for (int r = 0; r < 8; ++r) {
                    const int   idx   = (nt << 3) + r;
                    const float I64   = c[r];
                    const float vprev = vst[m][idx];
                    const float aprev = ast[m][idx];
                    const bool  so    = (spkm[m] >> idx) & 1u;
                    const float vk    = so ? 0.0f : vprev;
                    float u  = fmaf(oas_[nt], I64, -(oa_[nt] * aprev));
                    float vn = fmaf(al_[nt], vk, u);
                    const bool sn = vn > 1.0f;
                    ast[m][idx] = fmaf(rh_[nt], aprev, sn ? bt_[nt] : 0.0f);
                    vst[m][idx] = vn;
                    if (sn) nmask[m] |= (1u << idx);
                }
            }
        }

        __syncthreads();

#pragma unroll
        for (int m = 0; m < 2; ++m)
#pragma unroll
            for (int idx = 0; idx < 32; ++idx) {
                const int srow = (m << 4) + (half << 3) + (idx & 7);
                const int col  = (wv << 6) + ((idx >> 3) << 4) + lm;
                sl[srow * SP + col] =
                    ((nmask[m] >> idx) & 1u) ? (unsigned char)0x38
                                             : (unsigned char)0x00;
            }
        spkm[0] = nmask[0]; spkm[1] = nmask[1];
    }

    float* vbuf = (float*)smem;
#pragma unroll
    for (int m = 0; m < 2; ++m)
#pragma unroll
        for (int idx = 0; idx < 32; ++idx) {
            const int srow = (m << 4) + (half << 3) + (idx & 7);
            const int col  = (wv << 6) + ((idx >> 3) << 4) + lm;
            vbuf[srow * VPF + col] = vst[m][idx];
        }
    __syncthreads();

    if (tid < BT * NOUT) {
        const int bb = tid >> 1, o = tid & 1;
        const float* w2r = W2 + o * HID;
        float s = 0.0f;
        for (int h = 0; h < HID; ++h)
            s = fmaf(vbuf[bb * VPF + h], w2r[h], s);
        out[(b0 + bb) * NOUT + o] = s;
    }
}

extern "C" void kernel_launch(void* const* d_in, const int* in_sizes, int n_in,
                              void* d_out, int out_size, void* d_ws, size_t ws_size,
                              hipStream_t stream) {
    (void)in_sizes; (void)n_in; (void)out_size; (void)d_ws; (void)ws_size;
    const float* x     = (const float*)d_in[0];
    const float* W1    = (const float*)d_in[1];
    const float* Wrec  = (const float*)d_in[2];
    const float* W2    = (const float*)d_in[3];
    const float* alpha = (const float*)d_in[4];
    const float* rho   = (const float*)d_in[5];
    const float* beta  = (const float*)d_in[6];
    float* out = (float*)d_out;

    (void)hipFuncSetAttribute((const void*)adlif_scan_fp8_wmma,
                              hipFuncAttributeMaxDynamicSharedMemorySize,
                              (int)LDS_TOTAL);

    adlif_scan_fp8_wmma<<<dim3(B_TOT / BT), dim3(NTHR), LDS_TOTAL, stream>>>(
        x, W1, Wrec, W2, alpha, rho, beta, out);
}
